// MSA_24567212933618
// MI455X (gfx1250) — compile-verified, matrix operand order checked by reading, not run on hardware
//
#include <hip/hip_runtime.h>
#include <hip/hip_bf16.h>

typedef __attribute__((ext_vector_type(16))) _Float16 v16h;
typedef __attribute__((ext_vector_type(8)))  _Float16 v8h;
typedef __attribute__((ext_vector_type(8)))  float    v8f;

namespace msa {
constexpr int kB = 8, kS = 1024, kDim = 1024, kH = 16, kD = 64;
constexpr int kTile = 128;           // rows per workgroup tile
constexpr int kLdX = 72;             // padded LDS row (halves), 144B = 9*16B -> aligned & conflict-spread
constexpr int kLdP = 136;            // padded LDS row for P / Vt (272B = 17*16B)
}

__device__ __forceinline__ v8f wmma_f16(v16h a, v16h b, v8f c) {
  // v_wmma_f32_16x16x32_f16: D = A(16x32) x B(32x16) + C, f32 accum
  return __builtin_amdgcn_wmma_f32_16x16x32_f16(false, a, false, b, (short)0, c,
                                                false, false);
}

// A-fragment loader for 16x32 f16 tile at mat[(m)*ld + k], row-major source.
// ISA layout: lanes 0-15 -> row m=lane, K-octets {0-7,16-23};
//             lanes 16-31 -> row m=lane-16, K-octets {8-15,24-31}.
// Also serves as the B-operand gather when the B source is stored [N][K]
// row-major (i.e. B^T), since column n of B is then a contiguous row.
__device__ __forceinline__ v16h load_frag_rm(const _Float16* mat, int ld, int lane) {
  const int m  = lane & 15;
  const int kb = (lane & 16) ? 8 : 0;
  const _Float16* p = mat + m * ld + kb;
  v8h lo = *(const v8h*)(p);        // K = kb .. kb+7   (16B aligned: ld % 8 == 0)
  v8h hi = *(const v8h*)(p + 16);   // K = kb+16 .. kb+23
  return __builtin_shufflevector(lo, hi, 0, 1, 2, 3, 4, 5, 6, 7,
                                 8, 9, 10, 11, 12, 13, 14, 15);
}

// CDNA5 async copy: 16B per lane, global -> LDS, bypassing VGPRs (ASYNCcnt).
// Flat addresses of __shared__ objects carry the wave-relative LDS address in
// their low 32 bits (ISA 10.2).
__device__ __forceinline__ void async_copy_b128(const _Float16* g, _Float16* l) {
  unsigned int lds_addr = (unsigned int)(size_t)l;
  asm volatile("global_load_async_to_lds_b128 %0, %1, off"
               :: "v"(lds_addr), "v"(g)
               : "memory");
}

// ---------------------------------------------------------------------------
// Kernel 1: per-head QKV projection.  grid = (B*H, S/128), block = 256 (8 waves)
// q = 0.125*(xh @ Wq^T + bq), k = xh @ Wk^T + bk, v = xh @ Wv^T + bv
// Q,K stored f16 [B,H,S,D]; V stored transposed f16 [B,H,D,S].
// ---------------------------------------------------------------------------
__global__ __launch_bounds__(256) void msa_qkv_kernel(
    const float* __restrict__ x,  const float* __restrict__ Wq,
    const float* __restrict__ Wk, const float* __restrict__ Wv,
    const float* __restrict__ bq, const float* __restrict__ bk,
    const float* __restrict__ bv,
    _Float16* __restrict__ Qg, _Float16* __restrict__ Kg,
    _Float16* __restrict__ Vtg) {
  using namespace msa;
  __shared__ __attribute__((aligned(16))) _Float16 xs[kTile * kLdX];
  __shared__ __attribute__((aligned(16))) _Float16 wsm[3][kD * kLdX];

  const int bh = blockIdx.x;
  const int b = bh / kH, h = bh % kH;
  const int s0 = blockIdx.y * kTile;
  const int tid = threadIdx.x;
  const int wave = tid >> 5, lane = tid & 31;

  // stage x tile (cols h*64..h*64+63) as f16
  for (int i = tid; i < kTile * kD; i += 256) {
    const int r = i >> 6, d = i & 63;
    xs[r * kLdX + d] =
        (_Float16)x[((size_t)b * kS + (s0 + r)) * kDim + h * kD + d];
  }
  // stage the three 64x64 weight matrices for this head as f16 ([e][d] order)
  for (int i = tid; i < kD * kD; i += 256) {
    const size_t off = (size_t)h * kD * kD + i;
    const int e = i >> 6, d = i & 63;
    wsm[0][e * kLdX + d] = (_Float16)Wq[off];
    wsm[1][e * kLdX + d] = (_Float16)Wk[off];
    wsm[2][e * kLdX + d] = (_Float16)Wv[off];
  }
  __syncthreads();

  const int m0 = wave * 16;  // this wave's 16 rows
  const v16h a0 = load_frag_rm(xs + m0 * kLdX + 0,  kLdX, lane);
  const v16h a1 = load_frag_rm(xs + m0 * kLdX + 32, kLdX, lane);

  const int n_in = lane & 15;
  const int mofs = (lane & 16) ? 8 : 0;
  const size_t qkbase = ((size_t)bh * kS + s0 + m0) * kD;

  for (int proj = 0; proj < 3; ++proj) {
    const float* bias = (proj == 0) ? bq : (proj == 1) ? bk : bv;
#pragma unroll
    for (int nt = 0; nt < 4; ++nt) {
      // B = W^T gather: W stored [e][d] -> contiguous row per output column e
      const v16h b0 = load_frag_rm(&wsm[proj][(nt * 16) * kLdX + 0],  kLdX, lane);
      const v16h b1 = load_frag_rm(&wsm[proj][(nt * 16) * kLdX + 32], kLdX, lane);
      v8f acc = {};
      acc = wmma_f16(a0, b0, acc);
      acc = wmma_f16(a1, b1, acc);

      const int n = nt * 16 + n_in;
      const float bias_n = bias[h * kD + n];
#pragma unroll
      for (int i = 0; i < 8; ++i) acc[i] += bias_n;
      if (proj == 0) {
#pragma unroll
        for (int i = 0; i < 8; ++i) acc[i] *= 0.125f;  // 1/sqrt(D) folded into Q
      }

      if (proj == 2) {
        // V transposed: Vt[bh][n(d)][s] ; per lane 8 consecutive s positions
        const size_t vb = ((size_t)bh * kD + n) * kS + s0 + m0 + mofs;
#pragma unroll
        for (int i = 0; i < 8; ++i) Vtg[vb + i] = (_Float16)acc[i];
      } else {
        _Float16* out = (proj == 0) ? Qg : Kg;
#pragma unroll
        for (int i = 0; i < 8; ++i)
          out[qkbase + (size_t)(mofs + i) * kD + n] = (_Float16)acc[i];
      }
    }
  }
}

// ---------------------------------------------------------------------------
// Kernel 2: flash attention.  grid = (B*H, S/128), block = 256 (8 waves).
// Each wave owns 16 query rows; K/Vt tiles of 128 kv-positions stream into a
// double-buffered LDS pipeline via GLOBAL_LOAD_ASYNC_TO_LDS_B128 (ASYNCcnt).
// ---------------------------------------------------------------------------
__global__ __launch_bounds__(256) void msa_attn_kernel(
    const _Float16* __restrict__ Qg, const _Float16* __restrict__ Kg,
    const _Float16* __restrict__ Vtg, float* __restrict__ out) {
  using namespace msa;
  __shared__ __attribute__((aligned(16))) _Float16 qs[kTile * kLdX];
  __shared__ __attribute__((aligned(16))) _Float16 ks[2][kTile * kLdX];
  __shared__ __attribute__((aligned(16))) _Float16 vts[2][kD * kLdP];
  __shared__ __attribute__((aligned(16))) _Float16 ps[8 * 16 * kLdP];

  const int bh = blockIdx.x;
  const int b = bh / kH, h = bh % kH;
  const int s0 = blockIdx.y * kTile;
  const int tid = threadIdx.x;
  const int wave = tid >> 5, lane = tid & 31;

  const size_t qkbase = (size_t)bh * kS * kD;   // [B,H,S,D]
  const size_t vtbase = (size_t)bh * kD * kS;   // [B,H,D,S]

  // ---- async-stage the Q tile [128][64] ----
#pragma unroll
  for (int it = 0; it < 4; ++it) {
    const int idx = it * 2048 + tid * 8;       // 8 halves = 16B per op
    const int r = idx >> 6, d = idx & 63;
    async_copy_b128(Qg + qkbase + (size_t)(s0 + r) * kD + d, qs + r * kLdX + d);
  }
  asm volatile("s_wait_asynccnt 0x0" ::: "memory");
  __syncthreads();

  const int m0 = wave * 16;
  const v16h aq0 = load_frag_rm(qs + m0 * kLdX + 0,  kLdX, lane);
  const v16h aq1 = load_frag_rm(qs + m0 * kLdX + 32, kLdX, lane);

  // per-lane row state; VGPR slot i of every C/D tile maps to the same row
  float mrow[8], lrow[8];
  v8f o[4];
  const v8f vzero = {};
#pragma unroll
  for (int i = 0; i < 8; ++i) { mrow[i] = -3.0e38f; lrow[i] = 0.0f; }
#pragma unroll
  for (int nt = 0; nt < 4; ++nt) o[nt] = vzero;

  _Float16* pw = ps + wave * 16 * kLdP;  // this wave's private P strip
  const int n_in = lane & 15;
  const int mofs = (lane & 16) ? 8 : 0;

  // issue a kv tile (K [128][64] + Vt [64][128]) into LDS buffer `buf`
  auto issue_kv_tile = [&](int j0, int buf) {
    const _Float16* kg = Kg + qkbase + (size_t)j0 * kD;
    const _Float16* vg = Vtg + vtbase + j0;
    _Float16* kd = ks[buf];
    _Float16* vd = vts[buf];
#pragma unroll
    for (int it = 0; it < 4; ++it) {
      const int idx = it * 2048 + tid * 8;
      const int r = idx >> 6, d = idx & 63;
      async_copy_b128(kg + (size_t)r * kD + d, kd + r * kLdX + d);
    }
#pragma unroll
    for (int it = 0; it < 4; ++it) {
      const int idx = it * 2048 + tid * 8;
      const int d = idx >> 7, j = idx & 127;
      async_copy_b128(vg + (size_t)d * kS + j, vd + d * kLdP + j);
    }
  };

  issue_kv_tile(0, 0);  // prime the pipeline

  for (int t = 0; t < 8; ++t) {
    const int cur = t & 1;
    if (t < 7) {
      issue_kv_tile((t + 1) * 128, cur ^ 1);  // stream next tile behind compute
      // 8 ops of tile t may still be pending below 8 newly issued ones;
      // in-order completion => waiting to <=8 drains all of tile t.
      asm volatile("s_wait_asynccnt 0x8" ::: "memory");
    } else {
      asm volatile("s_wait_asynccnt 0x0" ::: "memory");
    }
    __syncthreads();

    const _Float16* kcur = ks[cur];
    const _Float16* vcur = vts[cur];

    // ---- scores: S = Q*K^T (K stored [j][d] row-major == B^T source) ----
    v8f sc[8];
#pragma unroll
    for (int ct = 0; ct < 8; ++ct) {
      const v16h b0 = load_frag_rm(kcur + (ct * 16) * kLdX + 0,  kLdX, lane);
      const v16h b1 = load_frag_rm(kcur + (ct * 16) * kLdX + 32, kLdX, lane);
      v8f acc = {};
      acc = wmma_f16(aq0, b0, acc);
      acc = wmma_f16(aq1, b1, acc);
      sc[ct] = acc;
    }

    // ---- online softmax: row max over the 128 columns ----
    float rmax[8];
#pragma unroll
    for (int i = 0; i < 8; ++i) rmax[i] = -3.0e38f;
#pragma unroll
    for (int ct = 0; ct < 8; ++ct)
#pragma unroll
      for (int i = 0; i < 8; ++i) rmax[i] = fmaxf(rmax[i], sc[ct][i]);
#pragma unroll
    for (int i = 0; i < 8; ++i) {  // reduce across the 16-lane N group
      float v = rmax[i];
      v = fmaxf(v, __shfl_xor(v, 1));
      v = fmaxf(v, __shfl_xor(v, 2));
      v = fmaxf(v, __shfl_xor(v, 4));
      v = fmaxf(v, __shfl_xor(v, 8));
      rmax[i] = v;
    }
#pragma unroll
    for (int i = 0; i < 8; ++i) {
      const float mn = fmaxf(mrow[i], rmax[i]);
      const float scale = __expf(mrow[i] - mn);
      mrow[i] = mn;
      lrow[i] *= scale;
#pragma unroll
      for (int nt = 0; nt < 4; ++nt) o[nt][i] *= scale;
    }

    // ---- P = exp(S - m), row sums, stage P (f16) for the PV matmul ----
#pragma unroll
    for (int ct = 0; ct < 8; ++ct) {
#pragma unroll
      for (int i = 0; i < 8; ++i) {
        const float p = __expf(sc[ct][i] - mrow[i]);
        sc[ct][i] = p;
        pw[(mofs + i) * kLdP + ct * 16 + n_in] = (_Float16)p;
      }
#pragma unroll
      for (int i = 0; i < 8; ++i) {
        float v = sc[ct][i];
        v += __shfl_xor(v, 1);
        v += __shfl_xor(v, 2);
        v += __shfl_xor(v, 4);
        v += __shfl_xor(v, 8);
        lrow[i] += v;
      }
    }

    // ---- O += P * V  (Vt stored [d][j] row-major == B^T source) ----
#pragma unroll
    for (int kc = 0; kc < 4; ++kc) {
      const v16h ap = load_frag_rm(pw + kc * 32, kLdP, lane);
#pragma unroll
      for (int nt = 0; nt < 4; ++nt) {
        const v16h bv = load_frag_rm(vcur + (nt * 16) * kLdP + kc * 32, kLdP, lane);
        o[nt] = wmma_f16(ap, bv, o[nt]);
      }
    }
    __syncthreads();  // all waves done with buffer `cur` before it is re-filled
  }

  // ---- epilogue: out[b, s0+m0+m, h*64+n] = O / l ----
#pragma unroll
  for (int nt = 0; nt < 4; ++nt) {
    const int n = nt * 16 + n_in;
#pragma unroll
    for (int i = 0; i < 8; ++i) {
      out[((size_t)b * kS + s0 + m0 + mofs + i) * kDim + h * kD + n] =
          o[nt][i] / lrow[i];
    }
  }
}

extern "C" void kernel_launch(void* const* d_in, const int* in_sizes, int n_in,
                              void* d_out, int out_size, void* d_ws,
                              size_t ws_size, hipStream_t stream) {
  using namespace msa;
  const float* x  = (const float*)d_in[0];
  const float* Wq = (const float*)d_in[1];
  const float* Wk = (const float*)d_in[2];
  const float* Wv = (const float*)d_in[3];
  const float* bq = (const float*)d_in[4];
  const float* bk = (const float*)d_in[5];
  const float* bv = (const float*)d_in[6];
  float* out = (float*)d_out;

  // workspace: Q,K f16 [B,H,S,D]; Vt f16 [B,H,D,S]  (3 * 16 MB = 48 MB)
  const size_t elems = (size_t)kB * kH * kS * kD;
  _Float16* Qg  = (_Float16*)d_ws;
  _Float16* Kg  = Qg + elems;
  _Float16* Vtg = Kg + elems;

  dim3 grid(kB * kH, kS / kTile);
  msa_qkv_kernel<<<grid, 256, 0, stream>>>(x, Wq, Wk, Wv, bq, bk, bv, Qg, Kg, Vtg);
  msa_attn_kernel<<<grid, 256, 0, stream>>>(Qg, Kg, Vtg, out);
}
